// AgentNet_26414048870991
// MI455X (gfx1250) — matrix operand order repaired and checked by reading, not compiled, not run on hardware
//
#include <hip/hip_runtime.h>
#include <math.h>

#define NN      16384
#define DEGK    32
#define FF      128
#define DDIM    64
#define PSTEPS  8
#define CAND    33

typedef __attribute__((ext_vector_type(16))) __bf16 v16bf;
typedef __attribute__((ext_vector_type(8)))  __bf16 v8bf;
typedef __attribute__((ext_vector_type(8)))  float  v8f;

static __device__ __forceinline__ v8f vzero8(){
    v8f z = {0.f,0.f,0.f,0.f,0.f,0.f,0.f,0.f};
    return z;
}
static __device__ __forceinline__ float lrelu(float x, float s){ return x > 0.f ? x : s * x; }

// ---------------- WMMA tile helpers (wave32; one wave computes a 16-row tile) -------------
// A rows staged in LDS as bf16 row-major [16][KTOT].
// A 16x32 layout (matches B): lanes 0-15 hold M=lane, K = k0+{0..15};
//                            lanes 16-31 hold M=lane-16, K = k0+{16..31}.
static __device__ __forceinline__ v16bf load_a_frag(const __bf16* sA, int ktot, int k0, int lane){
    int m  = lane & 15;
    int hi = (lane >> 4) & 1;
    const __bf16* r = sA + m * ktot + k0 + hi * 16;
    v8bf lo = *(const v8bf*)(r);
    v8bf hv = *(const v8bf*)(r + 8);
    v16bf a;
    #pragma unroll
    for (int i = 0; i < 8; ++i){ a[i] = lo[i]; a[8+i] = hv[i]; }
    return a;
}
// Wt: bf16 [NCOL][KTOT] (transposed weights). B 32x16 layout, same k order as A:
// lane holds column N=(lane&15)+cg*16, K = k0 + (lane>=16 ? 16 : 0) .. +15 (contiguous).
static __device__ __forceinline__ v16bf load_b_frag(const __bf16* Wt, int ktot, int k0, int cg, int lane){
    int n  = (lane & 15) + cg * 16;
    int ko = k0 + (((lane >> 4) & 1) ? 16 : 0);
    return *(const v16bf*)(Wt + (long)n * ktot + ko);
}

template<int KTOT>
static __device__ __forceinline__ void gemm_cols64(const __bf16* sA, const __bf16* Wt, v8f acc[4], int lane){
    #pragma unroll
    for (int k0 = 0; k0 < KTOT; k0 += 32){
        v16bf a = load_a_frag(sA, KTOT, k0, lane);
        #pragma unroll
        for (int cg = 0; cg < 4; ++cg){
            v16bf b = load_b_frag(Wt, KTOT, k0, cg, lane);
            acc[cg] = __builtin_amdgcn_wmma_f32_16x16x32_bf16(false, a, false, b, (short)0, acc[cg], false, false);
        }
    }
}
static __device__ __forceinline__ void gemm_cols128(const __bf16* sA, const __bf16* Wt, v8f acc[8], int lane){
    #pragma unroll
    for (int k0 = 0; k0 < 128; k0 += 32){
        v16bf a = load_a_frag(sA, 128, k0, lane);
        #pragma unroll
        for (int cg = 0; cg < 8; ++cg){
            v16bf b = load_b_frag(Wt, 128, k0, cg, lane);
            acc[cg] = __builtin_amdgcn_wmma_f32_16x16x32_bf16(false, a, false, b, (short)0, acc[cg], false, false);
        }
    }
}

// ---------------- setup kernels -------------------------------------------------------------

// Convert f32 weight [K][NC] -> bf16 transposed Wt[NC][K]
__global__ void k_wt(const float* __restrict__ W, __bf16* __restrict__ Wt, int K, int NC){
    int i = blockIdx.x * 256 + threadIdx.x;
    if (i < K * NC){
        int k = i / NC, n = i % NC;
        Wt[(long)n * K + k] = (__bf16)W[i];
    }
}

// Per-step time vectors: lte = LR(LR(table[s]@t1+b)@t2+b, .01); {ct,nt,at}v = lte@{ct,nt,at}_W + b
__global__ __launch_bounds__(128) void k_time(
    const float* __restrict__ ttab, const float* __restrict__ t1W, const float* __restrict__ t1b,
    const float* __restrict__ t2W, const float* __restrict__ t2b,
    const float* __restrict__ ctW, const float* __restrict__ ctb,
    const float* __restrict__ ntW, const float* __restrict__ ntb,
    const float* __restrict__ atW, const float* __restrict__ atb,
    float* __restrict__ ctv, float* __restrict__ ntv, float* __restrict__ atv)
{
    __shared__ float sh[DDIM];
    __shared__ float slt[DDIM];
    int t = threadIdx.x;
    for (int s = 1; s <= PSTEPS; ++s){
        const float* row = ttab + s * DDIM;
        if (t < DDIM){
            float a = t1b[t];
            for (int i = 0; i < DDIM; ++i) a += row[i] * t1W[i*DDIM + t];
            sh[t] = lrelu(a, 0.01f);
        }
        __syncthreads();
        if (t < DDIM){
            float a = t2b[t];
            for (int i = 0; i < DDIM; ++i) a += sh[i] * t2W[i*DDIM + t];
            slt[t] = lrelu(a, 0.01f);               // LR(te, 0.01)
        }
        __syncthreads();
        {
            float c = ctb[t], nv = ntb[t], av = atb[t];
            for (int i = 0; i < DDIM; ++i){
                float l = slt[i];
                c  += l * ctW[i*128 + t];
                nv += l * ntW[i*128 + t];
                av += l * atW[i*128 + t];
            }
            ctv[(s-1)*128 + t] = c;
            ntv[(s-1)*128 + t] = nv;
            atv[(s-1)*128 + t] = av;
        }
        __syncthreads();
    }
}

__global__ void k_init(const float* __restrict__ agent0, float* __restrict__ agent_emb,
                       int* __restrict__ cur, float* __restrict__ out){
    int i = blockIdx.x * 256 + threadIdx.x;
    if (i < NN){
        cur[i] = i;
        out[(long)i * (PSTEPS+1)] = (float)i;          // pattern column 0 = start node
        for (int j = 0; j < DDIM; ++j) agent_emb[(long)i*DDIM + j] = agent0[j];
    }
}

// node_emb = x @ W_in + b_in  (no LN)
__global__ __launch_bounds__(32) void k_node_init(const float* __restrict__ x, const __bf16* __restrict__ Wt,
                                                  const float* __restrict__ b, float* __restrict__ node_emb){
    __shared__ __attribute__((aligned(32))) __bf16 sA[16*128];
    int lane = threadIdx.x, m = lane & 15, half = lane >> 4;
    long r = (long)blockIdx.x * 16 + m;
    const float* src = x + r * FF + half * 64;
    #pragma unroll
    for (int j = 0; j < 64; ++j) sA[m*128 + half*64 + j] = (__bf16)src[j];
    __syncthreads();
    v8f acc[4] = {vzero8(),vzero8(),vzero8(),vzero8()};
    gemm_cols64<128>(sA, Wt, acc, lane);
    #pragma unroll
    for (int f = 0; f < 4; ++f)
        #pragma unroll
        for (int v = 0; v < 8; ++v){
            int mm = v + 8*half, col = (lane & 15) + f*16;
            node_emb[((long)blockIdx.x*16 + mm)*DDIM + col] = acc[f][v] + b[col];
        }
}

// ---------------- per-step kernels ----------------------------------------------------------

// q = LN(agent_emb; q_g,q_beta) @ q_W + q_b
__global__ __launch_bounds__(32) void k_q(const float* __restrict__ agent_emb,
    const float* __restrict__ g, const float* __restrict__ beta,
    const __bf16* __restrict__ Wt, const float* __restrict__ bias, float* __restrict__ qout)
{
    __shared__ __attribute__((aligned(32))) __bf16 sA[16*64];
    int lane = threadIdx.x, m = lane & 15, half = lane >> 4;
    long r = (long)blockIdx.x * 16 + m;
    const float* src = agent_emb + r*DDIM + half*32;
    float vals[32]; float s = 0.f, ss = 0.f;
    #pragma unroll
    for (int j = 0; j < 32; ++j){ float v = src[j]; vals[j] = v; s += v; ss += v*v; }
    s += __shfl_xor(s, 16, 32); ss += __shfl_xor(ss, 16, 32);
    float mean = s * (1.f/64.f);
    float rstd = rsqrtf(ss * (1.f/64.f) - mean*mean + 1e-5f);
    #pragma unroll
    for (int j = 0; j < 32; ++j){
        int cix = half*32 + j;
        sA[m*64 + cix] = (__bf16)((vals[j]-mean)*rstd*g[cix] + beta[cix]);
    }
    __syncthreads();
    v8f acc[4] = {vzero8(),vzero8(),vzero8(),vzero8()};
    gemm_cols64<64>(sA, Wt, acc, lane);
    #pragma unroll
    for (int f = 0; f < 4; ++f)
        #pragma unroll
        for (int v = 0; v < 8; ++v){
            int mm = v + 8*half, col = (lane & 15) + f*16;
            qout[((long)blockIdx.x*16 + mm)*DDIM + col] = acc[f][v] + bias[col];
        }
}

// rows r = n*33+c: k = LN(concat(node_emb[cand], node_emb[cur[n]])) @ k_W + k_b; logit = q[n].k / 8
__global__ __launch_bounds__(32) void k_attn(const float* __restrict__ node_emb, const float* __restrict__ q,
    const int* __restrict__ cur, const int* __restrict__ neigh,
    const float* __restrict__ g, const float* __restrict__ beta,
    const __bf16* __restrict__ Wt, const float* __restrict__ bias,
    float* __restrict__ logits)
{
    __shared__ __attribute__((aligned(32))) __bf16 sA[16*128];
    __shared__ float sK[16*64];
    int lane = threadIdx.x, m = lane & 15, half = lane >> 4;
    long r = (long)blockIdx.x * 16 + m;
    int n = (int)(r / CAND), c = (int)(r % CAND);
    int curn = cur[n];
    int cand = (c == 0) ? curn : neigh[(long)curn*DEGK + (c-1)];
    const float* src = node_emb + (long)(half ? curn : cand) * DDIM;
    int base = half * 64;
    float vals[64]; float s = 0.f, ss = 0.f;
    #pragma unroll
    for (int j = 0; j < 64; ++j){ float v = src[j]; vals[j] = v; s += v; ss += v*v; }
    s += __shfl_xor(s, 16, 32); ss += __shfl_xor(ss, 16, 32);
    float mean = s * (1.f/128.f);
    float rstd = rsqrtf(ss * (1.f/128.f) - mean*mean + 1e-5f);
    #pragma unroll
    for (int j = 0; j < 64; ++j){
        int cix = base + j;
        sA[m*128 + cix] = (__bf16)((vals[j]-mean)*rstd*g[cix] + beta[cix]);
    }
    __syncthreads();
    v8f acc[4] = {vzero8(),vzero8(),vzero8(),vzero8()};
    gemm_cols64<128>(sA, Wt, acc, lane);
    #pragma unroll
    for (int f = 0; f < 4; ++f)
        #pragma unroll
        for (int v = 0; v < 8; ++v){
            int mm = v + 8*half, col = (lane & 15) + f*16;
            sK[mm*64 + col] = acc[f][v] + bias[col];
        }
    __syncthreads();
    const float* qr = q + (long)n * DDIM;
    int cb = half * 32;
    float p = 0.f;
    #pragma unroll
    for (int j = 0; j < 32; ++j) p += sK[m*64 + cb + j] * qr[cb + j];
    p += __shfl_xor(p, 16, 32);
    if (half == 0) logits[r] = p * 0.125f;   // 1/sqrt(64)
}

// gumbel argmax + log_softmax at chosen candidate; writes nxt, pattern col, logp col
__global__ void k_sample(const float* __restrict__ logits, const float* __restrict__ u,
    const int* __restrict__ neigh, const int* __restrict__ cur, int* __restrict__ nxt,
    float* __restrict__ out, int step)
{
    int n = blockIdx.x * 256 + threadIdx.x;
    if (n >= NN) return;
    const float* lg = logits + (long)n * CAND;
    const float* un = u + (long)n * CAND;
    float best = -3.0e38f; int bc = 0; float lmax = -3.0e38f;
    for (int c = 0; c < CAND; ++c){
        float l = lg[c];
        lmax = fmaxf(lmax, l);
        float gb = -__logf(-__logf(un[c]));
        float v = l + gb;
        if (v > best){ best = v; bc = c; }
    }
    float sum = 0.f;
    for (int c = 0; c < CAND; ++c) sum += __expf(lg[c] - lmax);
    float logp = lg[bc] - lmax - __logf(sum);
    int curn = cur[n];
    int nx = (bc == 0) ? curn : neigh[(long)curn*DEGK + (bc-1)];
    nxt[n] = nx;
    out[(long)n*(PSTEPS+1) + step] = (float)nx;
    out[(long)NN*(PSTEPS+1) + (long)n*PSTEPS + (step-1)] = logp;
}

// agg[n] = sum_c LR(LN(node_emb[cand_all[n,c]]) @ msg_W + msg_b, 0.2)
__global__ __launch_bounds__(32) void k_msg_agg(const float* __restrict__ node_emb, const int* __restrict__ neigh,
    const float* __restrict__ g, const float* __restrict__ beta,
    const __bf16* __restrict__ Wt, const float* __restrict__ bias,
    float* __restrict__ agg)
{
    __shared__ __attribute__((aligned(32))) __bf16 sA[16*64];
    int lane = threadIdx.x, m = lane & 15, half = lane >> 4;
    int n = blockIdx.x;
    float sums[4][8];
    #pragma unroll
    for (int f = 0; f < 4; ++f)
        #pragma unroll
        for (int v = 0; v < 8; ++v) sums[f][v] = 0.f;
    for (int t = 0; t < 3; ++t){
        int c = t*16 + m;
        bool valid = (c < CAND);
        int cand = valid ? ((c == 0) ? n : neigh[(long)n*DEGK + (c-1)]) : 0;
        const float* src = node_emb + (long)cand*DDIM + half*32;
        float vals[32]; float s = 0.f, ss = 0.f;
        #pragma unroll
        for (int j = 0; j < 32; ++j){ float v = valid ? src[j] : 0.f; vals[j] = v; s += v; ss += v*v; }
        s += __shfl_xor(s, 16, 32); ss += __shfl_xor(ss, 16, 32);
        float mean = s * (1.f/64.f);
        float rstd = rsqrtf(ss * (1.f/64.f) - mean*mean + 1e-5f);
        #pragma unroll
        for (int j = 0; j < 32; ++j){
            int cix = half*32 + j;
            sA[m*64 + cix] = (__bf16)((vals[j]-mean)*rstd*g[cix] + beta[cix]);
        }
        __syncthreads();
        v8f acc[4] = {vzero8(),vzero8(),vzero8(),vzero8()};
        gemm_cols64<64>(sA, Wt, acc, lane);
        #pragma unroll
        for (int f = 0; f < 4; ++f)
            #pragma unroll
            for (int v = 0; v < 8; ++v){
                int rowc = t*16 + v + 8*half;
                if (rowc < CAND){
                    int col = (lane & 15) + f*16;
                    sums[f][v] += lrelu(acc[f][v] + bias[col], 0.2f);
                }
            }
        __syncthreads();
    }
    #pragma unroll
    for (int f = 0; f < 4; ++f){
        float p = 0.f;
        #pragma unroll
        for (int v = 0; v < 8; ++v) p += sums[f][v];
        p += __shfl_xor(p, 16, 32);
        if (half == 0) agg[(long)n*DDIM + (lane & 15) + f*16] = p;
    }
}

// conv MLP: node_emb += W2( LR(W1( LN(concat(node_emb,agg)+ct) ) + b1) ) + b2
__global__ __launch_bounds__(32) void k_conv(float* __restrict__ node_emb, const float* __restrict__ agg,
    const float* __restrict__ tvec,
    const float* __restrict__ g, const float* __restrict__ beta,
    const __bf16* __restrict__ W1t, const float* __restrict__ b1,
    const __bf16* __restrict__ W2t, const float* __restrict__ b2)
{
    __shared__ __attribute__((aligned(32))) __bf16 sA[16*128];
    __shared__ __attribute__((aligned(32))) __bf16 sH[16*128];
    int lane = threadIdx.x, m = lane & 15, half = lane >> 4;
    long r = (long)blockIdx.x * 16 + m;
    const float* src = half ? (agg + r*DDIM) : (node_emb + r*DDIM);
    int base = half * 64;
    float vals[64]; float s = 0.f, ss = 0.f;
    #pragma unroll
    for (int j = 0; j < 64; ++j){ float v = src[j] + tvec[base + j]; vals[j] = v; s += v; ss += v*v; }
    s += __shfl_xor(s, 16, 32); ss += __shfl_xor(ss, 16, 32);
    float mean = s * (1.f/128.f);
    float rstd = rsqrtf(ss * (1.f/128.f) - mean*mean + 1e-5f);
    #pragma unroll
    for (int j = 0; j < 64; ++j){
        int cix = base + j;
        sA[m*128 + cix] = (__bf16)((vals[j]-mean)*rstd*g[cix] + beta[cix]);
    }
    __syncthreads();
    v8f a1[8] = {vzero8(),vzero8(),vzero8(),vzero8(),vzero8(),vzero8(),vzero8(),vzero8()};
    gemm_cols128(sA, W1t, a1, lane);
    #pragma unroll
    for (int f = 0; f < 8; ++f)
        #pragma unroll
        for (int v = 0; v < 8; ++v){
            int mm = v + 8*half, col = (lane & 15) + f*16;
            sH[mm*128 + col] = (__bf16)lrelu(a1[f][v] + b1[col], 0.01f);
        }
    __syncthreads();
    v8f a2[4] = {vzero8(),vzero8(),vzero8(),vzero8()};
    gemm_cols64<128>(sH, W2t, a2, lane);
    #pragma unroll
    for (int f = 0; f < 4; ++f)
        #pragma unroll
        for (int v = 0; v < 8; ++v){
            int mm = v + 8*half, col = (lane & 15) + f*16;
            node_emb[((long)blockIdx.x*16 + mm)*DDIM + col] += a2[f][v] + b2[col];
        }
}

// node update: h = MLP(LN(concat(node_src[nxt], agent_emb)+nt)); node_dst[nxt] += h (atomic scatter)
__global__ __launch_bounds__(32) void k_node_upd(const float* __restrict__ node_src, float* __restrict__ node_dst,
    const float* __restrict__ agent_emb, const int* __restrict__ nxt, const float* __restrict__ tvec,
    const float* __restrict__ g, const float* __restrict__ beta,
    const __bf16* __restrict__ W1t, const float* __restrict__ b1,
    const __bf16* __restrict__ W2t, const float* __restrict__ b2)
{
    __shared__ __attribute__((aligned(32))) __bf16 sA[16*128];
    __shared__ __attribute__((aligned(32))) __bf16 sH[16*128];
    __shared__ int sNxt[16];
    int lane = threadIdx.x, m = lane & 15, half = lane >> 4;
    long r = (long)blockIdx.x * 16 + m;
    int nx = nxt[r];
    if (half == 0) sNxt[m] = nx;
    const float* src = half ? (agent_emb + r*DDIM) : (node_src + (long)nx*DDIM);
    int base = half * 64;
    float vals[64]; float s = 0.f, ss = 0.f;
    #pragma unroll
    for (int j = 0; j < 64; ++j){ float v = src[j] + tvec[base + j]; vals[j] = v; s += v; ss += v*v; }
    s += __shfl_xor(s, 16, 32); ss += __shfl_xor(ss, 16, 32);
    float mean = s * (1.f/128.f);
    float rstd = rsqrtf(ss * (1.f/128.f) - mean*mean + 1e-5f);
    #pragma unroll
    for (int j = 0; j < 64; ++j){
        int cix = base + j;
        sA[m*128 + cix] = (__bf16)((vals[j]-mean)*rstd*g[cix] + beta[cix]);
    }
    __syncthreads();
    v8f a1[8] = {vzero8(),vzero8(),vzero8(),vzero8(),vzero8(),vzero8(),vzero8(),vzero8()};
    gemm_cols128(sA, W1t, a1, lane);
    #pragma unroll
    for (int f = 0; f < 8; ++f)
        #pragma unroll
        for (int v = 0; v < 8; ++v){
            int mm = v + 8*half, col = (lane & 15) + f*16;
            sH[mm*128 + col] = (__bf16)lrelu(a1[f][v] + b1[col], 0.01f);
        }
    __syncthreads();
    v8f a2[4] = {vzero8(),vzero8(),vzero8(),vzero8()};
    gemm_cols64<128>(sH, W2t, a2, lane);
    #pragma unroll
    for (int f = 0; f < 4; ++f)
        #pragma unroll
        for (int v = 0; v < 8; ++v){
            int mm = v + 8*half, col = (lane & 15) + f*16;
            atomicAdd(&node_dst[(long)sNxt[mm]*DDIM + col], a2[f][v] + b2[col]);
        }
}

// agent update: agent_emb += MLP(LN(concat(agent_emb, node_emb[nxt])+at))
__global__ __launch_bounds__(32) void k_agent_upd(const float* __restrict__ node_emb, float* __restrict__ agent_emb,
    const int* __restrict__ nxt, const float* __restrict__ tvec,
    const float* __restrict__ g, const float* __restrict__ beta,
    const __bf16* __restrict__ W1t, const float* __restrict__ b1,
    const __bf16* __restrict__ W2t, const float* __restrict__ b2)
{
    __shared__ __attribute__((aligned(32))) __bf16 sA[16*128];
    __shared__ __attribute__((aligned(32))) __bf16 sH[16*128];
    int lane = threadIdx.x, m = lane & 15, half = lane >> 4;
    long r = (long)blockIdx.x * 16 + m;
    int nx = nxt[r];
    const float* src = half ? (node_emb + (long)nx*DDIM) : (agent_emb + r*DDIM);
    int base = half * 64;
    float vals[64]; float s = 0.f, ss = 0.f;
    #pragma unroll
    for (int j = 0; j < 64; ++j){ float v = src[j] + tvec[base + j]; vals[j] = v; s += v; ss += v*v; }
    s += __shfl_xor(s, 16, 32); ss += __shfl_xor(ss, 16, 32);
    float mean = s * (1.f/128.f);
    float rstd = rsqrtf(ss * (1.f/128.f) - mean*mean + 1e-5f);
    #pragma unroll
    for (int j = 0; j < 64; ++j){
        int cix = base + j;
        sA[m*128 + cix] = (__bf16)((vals[j]-mean)*rstd*g[cix] + beta[cix]);
    }
    __syncthreads();
    v8f a1[8] = {vzero8(),vzero8(),vzero8(),vzero8(),vzero8(),vzero8(),vzero8(),vzero8()};
    gemm_cols128(sA, W1t, a1, lane);
    #pragma unroll
    for (int f = 0; f < 8; ++f)
        #pragma unroll
        for (int v = 0; v < 8; ++v){
            int mm = v + 8*half, col = (lane & 15) + f*16;
            sH[mm*128 + col] = (__bf16)lrelu(a1[f][v] + b1[col], 0.01f);
        }
    __syncthreads();
    v8f a2[4] = {vzero8(),vzero8(),vzero8(),vzero8()};
    gemm_cols64<128>(sH, W2t, a2, lane);
    #pragma unroll
    for (int f = 0; f < 4; ++f)
        #pragma unroll
        for (int v = 0; v < 8; ++v){
            int mm = v + 8*half, col = (lane & 15) + f*16;
            agent_emb[((long)blockIdx.x*16 + mm)*DDIM + col] += a2[f][v] + b2[col];
        }
}

// ---------------- host launch ----------------------------------------------------------------

extern "C" void kernel_launch(void* const* d_in, const int* in_sizes, int n_in,
                              void* d_out, int out_size, void* d_ws, size_t ws_size,
                              hipStream_t stream)
{
    (void)in_sizes; (void)n_in; (void)out_size; (void)ws_size;

    const float* x       = (const float*)d_in[0];
    const int*   neigh   = (const int*)d_in[1];
    const float* gumbel  = (const float*)d_in[3];    // d_in[2] node_time unused by reference
    const float* W_in    = (const float*)d_in[4];
    const float* b_in    = (const float*)d_in[5];
    const float* ttab    = (const float*)d_in[6];
    const float* t1W     = (const float*)d_in[7];
    const float* t1b     = (const float*)d_in[8];
    const float* t2W     = (const float*)d_in[9];
    const float* t2b     = (const float*)d_in[10];
    const float* agent0  = (const float*)d_in[11];
    const float* msg_g   = (const float*)d_in[12];
    const float* msg_bt  = (const float*)d_in[13];
    const float* msg_W   = (const float*)d_in[14];
    const float* msg_b   = (const float*)d_in[15];
    const float* q_g     = (const float*)d_in[16];
    const float* q_bt    = (const float*)d_in[17];
    const float* q_W     = (const float*)d_in[18];
    const float* q_b     = (const float*)d_in[19];
    const float* kk_g    = (const float*)d_in[20];
    const float* kk_bt   = (const float*)d_in[21];
    const float* kk_W    = (const float*)d_in[22];
    const float* kk_b    = (const float*)d_in[23];
    const float* node_g  = (const float*)d_in[24];
    const float* node_bt = (const float*)d_in[25];
    const float* node_W1 = (const float*)d_in[26];
    const float* node_b1 = (const float*)d_in[27];
    const float* node_W2 = (const float*)d_in[28];
    const float* node_b2 = (const float*)d_in[29];
    const float* conv_g  = (const float*)d_in[30];
    const float* conv_bt = (const float*)d_in[31];
    const float* conv_W1 = (const float*)d_in[32];
    const float* conv_b1 = (const float*)d_in[33];
    const float* conv_W2 = (const float*)d_in[34];
    const float* conv_b2 = (const float*)d_in[35];
    const float* ag_g    = (const float*)d_in[36];
    const float* ag_bt   = (const float*)d_in[37];
    const float* ag_W1   = (const float*)d_in[38];
    const float* ag_b1   = (const float*)d_in[39];
    const float* ag_W2   = (const float*)d_in[40];
    const float* ag_b2   = (const float*)d_in[41];
    const float* nt_W    = (const float*)d_in[42];
    const float* nt_b    = (const float*)d_in[43];
    const float* at_W    = (const float*)d_in[44];
    const float* at_b    = (const float*)d_in[45];
    const float* ct_W    = (const float*)d_in[46];
    const float* ct_b    = (const float*)d_in[47];

    float* out = (float*)d_out;

    // ---- carve workspace ----
    char* base = (char*)d_ws;
    size_t off = 0;
    auto carve = [&](size_t bytes)->char*{
        off = (off + 255) & ~(size_t)255;
        char* p = base + off; off += bytes; return p;
    };
    __bf16* WtIn  = (__bf16*)carve((size_t)64*128*2);
    __bf16* WtQ   = (__bf16*)carve((size_t)64*64*2);
    __bf16* WtK   = (__bf16*)carve((size_t)64*128*2);
    __bf16* WtMsg = (__bf16*)carve((size_t)64*64*2);
    __bf16* WtC1  = (__bf16*)carve((size_t)128*128*2);
    __bf16* WtC2  = (__bf16*)carve((size_t)64*128*2);
    __bf16* WtN1  = (__bf16*)carve((size_t)128*128*2);
    __bf16* WtN2  = (__bf16*)carve((size_t)64*128*2);
    __bf16* WtA1  = (__bf16*)carve((size_t)128*128*2);
    __bf16* WtA2  = (__bf16*)carve((size_t)64*128*2);
    float* ctv       = (float*)carve((size_t)PSTEPS*128*4);
    float* ntv       = (float*)carve((size_t)PSTEPS*128*4);
    float* atv       = (float*)carve((size_t)PSTEPS*128*4);
    float* node_emb  = (float*)carve((size_t)NN*DDIM*4);
    float* node_tmp  = (float*)carve((size_t)NN*DDIM*4);
    float* agent_emb = (float*)carve((size_t)NN*DDIM*4);
    float* qbuf      = (float*)carve((size_t)NN*DDIM*4);
    float* aggbuf    = (float*)carve((size_t)NN*DDIM*4);
    float* logits    = (float*)carve((size_t)NN*CAND*4);
    int*   cbuf0     = (int*)carve((size_t)NN*4);
    int*   cbuf1     = (int*)carve((size_t)NN*4);

    // ---- weight conversion to bf16 transposed ----
    struct WConv { const float* W; __bf16* Wt; int K; int NC; };
    WConv wl[10] = {
        { W_in,    WtIn,  128,  64 },
        { q_W,     WtQ,    64,  64 },
        { kk_W,    WtK,   128,  64 },
        { msg_W,   WtMsg,  64,  64 },
        { conv_W1, WtC1,  128, 128 },
        { conv_W2, WtC2,  128,  64 },
        { node_W1, WtN1,  128, 128 },
        { node_W2, WtN2,  128,  64 },
        { ag_W1,   WtA1,  128, 128 },
        { ag_W2,   WtA2,  128,  64 },
    };
    for (int i = 0; i < 10; ++i)
        k_wt<<<(wl[i].K*wl[i].NC + 255)/256, 256, 0, stream>>>(wl[i].W, wl[i].Wt, wl[i].K, wl[i].NC);

    k_time<<<1, 128, 0, stream>>>(ttab, t1W, t1b, t2W, t2b, ct_W, ct_b, nt_W, nt_b, at_W, at_b,
                                  ctv, ntv, atv);
    k_init<<<(NN + 255)/256, 256, 0, stream>>>(agent0, agent_emb, cbuf0, out);
    k_node_init<<<NN/16, 32, 0, stream>>>(x, WtIn, b_in, node_emb);

    int* cur = cbuf0;
    int* nxt = cbuf1;
    const long attn_tiles = ((long)NN * CAND) / 16;

    for (int step = 1; step <= PSTEPS; ++step){
        const float* ct_s = ctv + (size_t)(step-1)*128;
        const float* nt_s = ntv + (size_t)(step-1)*128;
        const float* at_s = atv + (size_t)(step-1)*128;
        const float* u_s  = gumbel + (size_t)(step-1)*NN*CAND;

        k_q<<<NN/16, 32, 0, stream>>>(agent_emb, q_g, q_bt, WtQ, q_b, qbuf);
        k_attn<<<(unsigned)attn_tiles, 32, 0, stream>>>(node_emb, qbuf, cur, neigh,
                                                        kk_g, kk_bt, WtK, kk_b, logits);
        k_sample<<<(NN + 255)/256, 256, 0, stream>>>(logits, u_s, neigh, cur, nxt, out, step);
        k_msg_agg<<<NN, 32, 0, stream>>>(node_emb, neigh, msg_g, msg_bt, WtMsg, msg_b, aggbuf);
        k_conv<<<NN/16, 32, 0, stream>>>(node_emb, aggbuf, ct_s, conv_g, conv_bt,
                                         WtC1, conv_b1, WtC2, conv_b2);
        // snapshot node_emb so the node-update gather reads pre-scatter values
        hipMemcpyAsync(node_tmp, node_emb, (size_t)NN*DDIM*4, hipMemcpyDeviceToDevice, stream);
        k_node_upd<<<NN/16, 32, 0, stream>>>(node_tmp, node_emb, agent_emb, nxt, nt_s,
                                             node_g, node_bt, WtN1, node_b1, WtN2, node_b2);
        k_agent_upd<<<NN/16, 32, 0, stream>>>(node_emb, agent_emb, nxt, at_s,
                                              ag_g, ag_bt, WtA1, ag_b1, WtA2, ag_b2);
        int* t = cur; cur = nxt; nxt = t;
    }
}
